// Attention_33131377722062
// MI455X (gfx1250) — matrix operand order repaired and checked by reading, not compiled, not run on hardware
//
#include <hip/hip_runtime.h>
#include <hip/hip_bf16.h>
#include <stdint.h>

typedef __attribute__((ext_vector_type(16))) __bf16 v16bf;
typedef __attribute__((ext_vector_type(8)))  float  v8f;
typedef __attribute__((ext_vector_type(4)))  unsigned int u32x4;

#define B_  4
#define S_  2048
#define D_  1024
#define H_  16
#define DH_ 64
#define M_  (B_ * S_)   // 8192 tokens
#define N3_ (3 * D_)    // 3072

__device__ __forceinline__ unsigned short f2bf(float f) {
    union { float f; unsigned int u; } x; x.f = f;
    unsigned int r = x.u + 0x7FFFu + ((x.u >> 16) & 1u);   // round-to-nearest-even
    return (unsigned short)(r >> 16);
}

union FragB16 { v16bf v; u32x4 u[2]; };

// A fragment, 16x32 bf16 row-major [M,K] storage.
// lane: m = lane&15 ; k = (lane>>4)*8 + {0..7} and + {16..23}
__device__ __forceinline__ v16bf load_frag_a(const unsigned short* base, int ld,
                                             int row0, int k0, int lane) {
    int m  = row0 + (lane & 15);
    int kh = (lane >> 4) * 8;
    const unsigned short* p = base + (size_t)m * ld + k0 + kh;
    FragB16 f;
    f.u[0] = *(const u32x4*)(p);
    f.u[1] = *(const u32x4*)(p + 16);
    return f.v;
}

// B fragment, 32x16 bf16, from transposed storage BT[N,K].
// lane: n = lane&15 ; k = (lane>>4)*8 + {0..7} and + {16..23}
__device__ __forceinline__ v16bf load_frag_b(const unsigned short* baseT, int ld,
                                             int n0, int k0, int lane) {
    int n  = n0 + (lane & 15);
    int kh = (lane >> 4) * 8;
    const unsigned short* p = baseT + (size_t)n * ld + k0 + kh;
    FragB16 f;
    f.u[0] = *(const u32x4*)(p);
    f.u[1] = *(const u32x4*)(p + 16);
    return f.v;
}

__device__ __forceinline__ v8f wmma_bf16(v16bf a, v16bf b, v8f c) {
    return __builtin_amdgcn_wmma_f32_16x16x32_bf16(false, a, false, b, (short)0, c,
                                                   false, false);
}

// ---------------------------------------------------------------- converts
__global__ void attn_cvt_bf16(const float* __restrict__ in,
                              unsigned short* __restrict__ out, int n) {
    int i = blockIdx.x * blockDim.x + threadIdx.x;
    if (i < n) out[i] = f2bf(in[i]);
}

// out[n*K + k] = bf16(in[k*N + n])   (make BT = [N,K])
__global__ void attn_cvt_transpose(const float* __restrict__ in,
                                   unsigned short* __restrict__ out, int K, int N) {
    int i = blockIdx.x * blockDim.x + threadIdx.x;
    if (i < K * N) {
        int n = i / K, k = i % K;
        out[i] = f2bf(in[k * N + n]);
    }
}

// ---------------------------------------------------------------- QKV GEMM
// [M,1024] x [1024,3072] + bias  -> scatter to q[BH,S,64], k[BH,S,64], vT[BH,64,S]
// 4 waves / block, each wave owns a 64x64 tile (16 WMMA per 32-deep K step).
__global__ __launch_bounds__(128) void attn_qkv_gemm(
    const unsigned short* __restrict__ xb,   // [M, D] bf16
    const unsigned short* __restrict__ waT,  // [3D, D] bf16 (transposed)
    const float* __restrict__ b_attn,        // [3D]
    unsigned short* __restrict__ qb,
    unsigned short* __restrict__ kb,
    unsigned short* __restrict__ vT) {
    const int lane = threadIdx.x & 31;
    const int wave = threadIdx.x >> 5;
    const int wm = wave >> 1, wn = wave & 1;
    const int mbase = blockIdx.x * 128 + wm * 64;
    const int nbase = blockIdx.y * 128 + wn * 64;
    const int lm = lane & 15, lh = lane >> 4;

    v8f acc[4][4] = {};
    for (int kk = 0; kk < D_; kk += 32) {
        v16bf af[4], bf[4];
#pragma unroll
        for (int i = 0; i < 4; ++i) af[i] = load_frag_a(xb, D_, mbase + i * 16, kk, lane);
#pragma unroll
        for (int j = 0; j < 4; ++j) bf[j] = load_frag_b(waT, D_, nbase + j * 16, kk, lane);
        if (kk + 256 < D_) {   // pull future K-tiles toward the WGP (global_prefetch_b8)
            __builtin_prefetch(xb  + (size_t)(mbase + lane) * D_ + kk + 256, 0, 1);
            __builtin_prefetch(waT + (size_t)(nbase + lane) * D_ + kk + 256, 0, 1);
        }
#pragma unroll
        for (int i = 0; i < 4; ++i)
#pragma unroll
            for (int j = 0; j < 4; ++j)
                acc[i][j] = wmma_bf16(af[i], bf[j], acc[i][j]);
    }

    for (int i = 0; i < 4; ++i)
        for (int j = 0; j < 4; ++j) {
            int n     = nbase + j * 16 + lm;
            int which = n >> 10;            // 0:q 1:k 2:v
            int d     = n & 1023;
            int h     = d >> 6, dh = d & 63;
            float bias = b_attn[n];
#pragma unroll
            for (int r = 0; r < 8; ++r) {
                int m = mbase + i * 16 + r + 8 * lh;
                int b = m >> 11, s = m & 2047;
                int bh = b * H_ + h;
                unsigned short val = f2bf(acc[i][j][r] + bias);
                if (which == 0)      qb[((size_t)bh * S_ + s) * DH_ + dh] = val;
                else if (which == 1) kb[((size_t)bh * S_ + s) * DH_ + dh] = val;
                else                 vT[((size_t)bh * DH_ + dh) * S_ + s] = val;
            }
        }
}

// ---------------------------------------------------------------- attention
// grid (S/64, H, B), 4 waves; each wave: 16 query rows, flash online softmax.
__global__ __launch_bounds__(128) void attn_flash(
    const unsigned short* __restrict__ qb,
    const unsigned short* __restrict__ kb,
    const unsigned short* __restrict__ vT,
    unsigned short* __restrict__ ab) {
    __shared__ unsigned short sP[4][16 * 32];
    const int lane = threadIdx.x & 31;
    const int wave = threadIdx.x >> 5;
    const int lm = lane & 15, lh = lane >> 4;
    const int h = blockIdx.y, b = blockIdx.z;
    const int bh = b * H_ + h;
    const int qrow0 = blockIdx.x * 64 + wave * 16;

    const unsigned short* qh = qb + (size_t)bh * S_ * DH_;
    const unsigned short* kh = kb + (size_t)bh * S_ * DH_;
    const unsigned short* vh = vT + (size_t)bh * DH_ * S_;

    const v16bf aQ0 = load_frag_a(qh, DH_, qrow0, 0,  lane);
    const v16bf aQ1 = load_frag_a(qh, DH_, qrow0, 32, lane);

    v8f accO[4] = {};
    float mrun[8], lrun[8];
#pragma unroll
    for (int r = 0; r < 8; ++r) { mrun[r] = -3.0e38f; lrun[r] = 0.f; }

    const float scale = 0.125f;            // 1/sqrt(64)
    const int kend = qrow0 + 16;           // last needed key + 1 for this wave
    unsigned short* slab = sP[wave];

    for (int kbk = 0; kbk < kend; kbk += 32) {
        // ---- scores: Q(16x64) . K^T(64x32) -> two 16x16 D tiles
        v8f sc[2] = {};
#pragma unroll
        for (int t = 0; t < 2; ++t) {
            v16bf bk0 = load_frag_b(kh, DH_, kbk + t * 16, 0,  lane);
            v16bf bk1 = load_frag_b(kh, DH_, kbk + t * 16, 32, lane);
            sc[t] = wmma_bf16(aQ0, bk0, sc[t]);
            sc[t] = wmma_bf16(aQ1, bk1, sc[t]);
        }
        // ---- scale + additive causal mask (-1e9 like the reference)
#pragma unroll
        for (int t = 0; t < 2; ++t)
#pragma unroll
            for (int r = 0; r < 8; ++r) {
                int qr  = qrow0 + r + 8 * lh;
                int key = kbk + t * 16 + lm;
                float s = sc[t][r] * scale;
                sc[t][r] = (key > qr) ? -1.0e9f : s;
            }
        // ---- online softmax (row reductions across 16 lanes of each half)
        float mnew[8], alpha[8];
#pragma unroll
        for (int r = 0; r < 8; ++r) {
            float v = fmaxf(sc[0][r], sc[1][r]);
            v = fmaxf(v, __shfl_xor(v, 1));
            v = fmaxf(v, __shfl_xor(v, 2));
            v = fmaxf(v, __shfl_xor(v, 4));
            v = fmaxf(v, __shfl_xor(v, 8));
            mnew[r]  = fmaxf(mrun[r], v);
            alpha[r] = __expf(mrun[r] - mnew[r]);
            mrun[r]  = mnew[r];
        }
#pragma unroll
        for (int t = 0; t < 2; ++t)
#pragma unroll
            for (int r = 0; r < 8; ++r)
                sc[t][r] = __expf(sc[t][r] - mnew[r]);
#pragma unroll
        for (int r = 0; r < 8; ++r) {
            float v = sc[0][r] + sc[1][r];
            v += __shfl_xor(v, 1);
            v += __shfl_xor(v, 2);
            v += __shfl_xor(v, 4);
            v += __shfl_xor(v, 8);
            lrun[r] = lrun[r] * alpha[r] + v;
        }
#pragma unroll
        for (int j = 0; j < 4; ++j)
#pragma unroll
            for (int r = 0; r < 8; ++r)
                accO[j][r] = accO[j][r] * alpha[r];

        // ---- P: D-layout f32 -> bf16 A-layout via per-wave LDS slab
#pragma unroll
        for (int t = 0; t < 2; ++t)
#pragma unroll
            for (int r = 0; r < 8; ++r)
                slab[(r + 8 * lh) * 32 + t * 16 + lm] = f2bf(sc[t][r]);
        // per-wave DS ops are in-order -> safe to read back immediately
        FragB16 aP;
        {
            const unsigned short* p = slab + lm * 32 + lh * 8;
            aP.u[0] = *(const u32x4*)(p);
            aP.u[1] = *(const u32x4*)(p + 16);
        }
        // ---- O += P(16x32) . V(32x64), V stored transposed [64,S]
#pragma unroll
        for (int j = 0; j < 4; ++j) {
            v16bf bv = load_frag_b(vh, S_, j * 16, kbk, lane);
            accO[j] = wmma_bf16(aP.v, bv, accO[j]);
        }
    }

    float rn[8];
#pragma unroll
    for (int r = 0; r < 8; ++r) rn[r] = 1.0f / lrun[r];
#pragma unroll
    for (int j = 0; j < 4; ++j)
#pragma unroll
        for (int r = 0; r < 8; ++r) {
            int qr  = qrow0 + r + 8 * lh;
            int col = h * DH_ + j * 16 + lm;
            ab[((size_t)(b * S_ + qr)) * D_ + col] = f2bf(accO[j][r] * rn[r]);
        }
}

// ---------------------------------------------------------------- out proj
// Same 64x64-per-wave register tiling, f32 output + bias straight to d_out.
__global__ __launch_bounds__(128) void attn_proj_gemm(
    const unsigned short* __restrict__ ab,   // [M, D] bf16
    const unsigned short* __restrict__ wpT,  // [D, D] bf16 (transposed)
    const float* __restrict__ b_proj,
    float* __restrict__ out) {
    const int lane = threadIdx.x & 31;
    const int wave = threadIdx.x >> 5;
    const int wm = wave >> 1, wn = wave & 1;
    const int mbase = blockIdx.x * 128 + wm * 64;
    const int nbase = blockIdx.y * 128 + wn * 64;
    const int lm = lane & 15, lh = lane >> 4;

    v8f acc[4][4] = {};
    for (int kk = 0; kk < D_; kk += 32) {
        v16bf af[4], bf[4];
#pragma unroll
        for (int i = 0; i < 4; ++i) af[i] = load_frag_a(ab, D_, mbase + i * 16, kk, lane);
#pragma unroll
        for (int j = 0; j < 4; ++j) bf[j] = load_frag_b(wpT, D_, nbase + j * 16, kk, lane);
        if (kk + 256 < D_) {
            __builtin_prefetch(ab  + (size_t)(mbase + lane) * D_ + kk + 256, 0, 1);
            __builtin_prefetch(wpT + (size_t)(nbase + lane) * D_ + kk + 256, 0, 1);
        }
#pragma unroll
        for (int i = 0; i < 4; ++i)
#pragma unroll
            for (int j = 0; j < 4; ++j)
                acc[i][j] = wmma_bf16(af[i], bf[j], acc[i][j]);
    }

    for (int i = 0; i < 4; ++i)
        for (int j = 0; j < 4; ++j) {
            int n = nbase + j * 16 + lm;
            float bias = b_proj[n];
#pragma unroll
            for (int r = 0; r < 8; ++r) {
                int m = mbase + i * 16 + r + 8 * lh;
                out[(size_t)m * D_ + n] = acc[i][j][r] + bias;
            }
        }
}

// ---------------------------------------------------------------- launch
extern "C" void kernel_launch(void* const* d_in, const int* in_sizes, int n_in,
                              void* d_out, int out_size, void* d_ws, size_t ws_size,
                              hipStream_t stream) {
    (void)in_sizes; (void)n_in; (void)out_size; (void)ws_size;
    const float* x      = (const float*)d_in[0];
    const float* w_attn = (const float*)d_in[1];
    const float* b_attn = (const float*)d_in[2];
    const float* w_proj = (const float*)d_in[3];
    const float* b_proj = (const float*)d_in[4];
    float* out = (float*)d_out;

    unsigned short* ws  = (unsigned short*)d_ws;          // bf16 workspace, ~88 MiB
    unsigned short* xb  = ws;                             // [M, D]
    unsigned short* waT = xb  + (size_t)M_ * D_;          // [3D, D]
    unsigned short* wpT = waT + (size_t)N3_ * D_;         // [D, D]
    unsigned short* qb  = wpT + (size_t)D_ * D_;          // [B*H, S, DH]
    unsigned short* kb  = qb  + (size_t)B_ * H_ * S_ * DH_;
    unsigned short* vT  = kb  + (size_t)B_ * H_ * S_ * DH_;   // [B*H, DH, S]
    unsigned short* ab  = vT  + (size_t)B_ * H_ * S_ * DH_;   // [M, D]

    const int nx = M_ * D_;
    attn_cvt_bf16<<<(nx + 255) / 256, 256, 0, stream>>>(x, xb, nx);
    attn_cvt_transpose<<<(N3_ * D_ + 255) / 256, 256, 0, stream>>>(w_attn, waT, D_, N3_);
    attn_cvt_transpose<<<(D_ * D_ + 255) / 256, 256, 0, stream>>>(w_proj, wpT, D_, D_);
    attn_qkv_gemm<<<dim3(M_ / 128, N3_ / 128), 128, 0, stream>>>(xb, waT, b_attn, qb, kb, vT);
    attn_flash<<<dim3(S_ / 64, H_, B_), 128, 0, stream>>>(qb, kb, vT, ab);
    attn_proj_gemm<<<dim3(M_ / 128, D_ / 128), 128, 0, stream>>>(ab, wpT, b_proj, out);
}
